// Decoder_35845797052933
// MI455X (gfx1250) — matrix operand order repaired and checked by reading, not compiled, not run on hardware
//
#include <hip/hip_runtime.h>
#include <math.h>

#define B_   128
#define S_   400
#define V_   50000
#define EXT_ 100
#define H_   256
#define E_   128
#define VE_  (V_ + EXT_)      // 50100
#define KAW  512              // 2H : K of attention GEMM
#define NH   256              // H  : N of attention GEMM

typedef __attribute__((ext_vector_type(16))) __bf16        v16bf;
typedef __attribute__((ext_vector_type(8)))  float         v8f;
typedef __attribute__((ext_vector_type(8)))  unsigned short u16x8;

union BfVec { v16bf v; u16x8 h[2]; };

// let the backend pick the native f32->bf16 convert (RNE)
__device__ __forceinline__ unsigned short f2bf(float x) {
  union { __bf16 b; unsigned short u; } cv;
  cv.b = (__bf16)x;
  return cv.u;
}
__device__ __forceinline__ float sigm(float x) { return 1.0f / (1.0f + expf(-x)); }

// -------- K1: embedding + LSTM cell + Pgen partial (emb.wordW + h.hiddenW + b) ----------
__global__ void lstm_kernel(const int* word, const float* hidden, const float* cell,
                            const float* embed, const float* W_ih, const float* W_hh,
                            const float* b_ih, const float* b_hh,
                            const float* hiddenW, const float* wordW, const float* bscal,
                            float* pgen_part, float* h_out, float* c_out) {
  __shared__ float esh[E_];
  __shared__ float hsh[H_];
  __shared__ float red[H_];
  int b = blockIdx.x, tid = threadIdx.x;
  int w = word[b];
  if (tid < E_) esh[tid] = embed[(size_t)w * E_ + tid];
  hsh[tid] = hidden[b * H_ + tid];
  __syncthreads();
  float g[4];
  #pragma unroll
  for (int q = 0; q < 4; ++q) {               // i, f, g, o rows: tid + q*256
    int n = tid + q * H_;
    float a = b_ih[n] + b_hh[n];
    const float* wi = W_ih + (size_t)n * E_;
    for (int k = 0; k < E_; ++k) a += esh[k] * wi[k];
    const float* wh = W_hh + (size_t)n * H_;
    for (int k = 0; k < H_; ++k) a += hsh[k] * wh[k];
    g[q] = a;
  }
  float c = sigm(g[1]) * cell[b * H_ + tid] + sigm(g[0]) * tanhf(g[2]);
  float h = tanhf(sigm(g[3]) * tanhf(c));     // note: extra tanh per reference
  h_out[b * H_ + tid] = h;
  c_out[b * H_ + tid] = c;
  float pp = h * hiddenW[tid];
  if (tid < E_) pp += esh[tid] * wordW[tid];
  red[tid] = pp; __syncthreads();
  for (int off = 128; off > 0; off >>= 1) { if (tid < off) red[tid] += red[tid + off]; __syncthreads(); }
  if (tid == 0) pgen_part[b] = red[0] + bscal[0];
}

// -------- K2: hpart[b][n] = attn_b[n] + h_t[b] . attn_W[n][512:768] ----------
__global__ void hpart_kernel(const float* attn_W, const float* attn_b, const float* h_t,
                             float* hpart) {
  __shared__ float hsh[H_];
  int b = blockIdx.x, tid = threadIdx.x;
  hsh[tid] = h_t[b * H_ + tid];
  __syncthreads();
  const float* w = attn_W + (size_t)tid * (3 * H_) + 2 * H_;
  float a = attn_b[tid];
  for (int k = 0; k < H_; ++k) a += hsh[k] * w[k];
  hpart[b * H_ + tid] = a;
}

// -------- K2b: one-time bf16 conversion of attn_We (256 x 512) into workspace ----------
// webf[n*512 + k] = bf16(attn_W[n*768 + k]); stays L2-resident (256 KB), re-read by 400 blocks.
__global__ void convWe_kernel(const float* attn_W, unsigned short* webf) {
  int i = blockIdx.x * 256 + threadIdx.x;     // i < 256*512
  int n = i >> 9, k = i & 511;
  webf[i] = f2bf(attn_W[(size_t)n * (3 * H_) + k]);
}

// -------- K3: WMMA attention scores: scores[b,s] = v . tanh(enc . We^T + hpart) ----------
// Block = 256 thr (8 waves); each wave owns a 16-row M-tile, all 16 N-tiles (N=256).
// bf16 attn_We staged into 128 KB LDS per K-phase with global_load_async_to_lds_b128.
__global__ void __launch_bounds__(256) scores_kernel(const float* enc, const unsigned short* webf,
                                                     const float* hpart, const float* vvec,
                                                     float* scores) {
  __shared__ unsigned short Bsh[NH * 256];    // [n*256 + k'], 128 KB, per K-phase
  int tid  = threadIdx.x;
  int wave = tid >> 5, lane = tid & 31;
  int hi = lane >> 4, lm = lane & 15;
  int ka0 = hi ? 8 : 0;                       // A-operand interleaved K pattern
  int kb0 = hi ? 16 : 0;                      // unused; B reads with ka0, same K order as A
  int rowbase = (blockIdx.x * 8 + wave) * 16; // 16 rows of (b,s); 400%16==0 -> single b
  int b = rowbase / S_;
  const float* arow = enc + (size_t)(rowbase + lm) * KAW;
  unsigned long long sbase = (unsigned long long)webf;

  v8f vz = {};
  v8f acc[16];
  #pragma unroll
  for (int t = 0; t < 16; ++t) acc[t] = vz;

  for (int phase = 0; phase < 2; ++phase) {
    __syncthreads();                          // previous-phase consumers done (waits DScnt)
    // async copy: 8192 x 16B chunks; chunk c: n = c>>5, q = c&31 (8 ushorts each)
    for (int c = tid; c < 8192; c += 256) {
      int n = c >> 5, q = c & 31;
      unsigned int voff = (unsigned int)(((n << 9) + (phase << 8) + (q << 3)) * 2);
      unsigned int dst  = (unsigned int)(size_t)&Bsh[(n << 8) + (q << 3)];
      asm volatile("global_load_async_to_lds_b128 %0, %1, %2"
                   :: "v"(dst), "v"(voff), "s"(sbase) : "memory");
    }
    asm volatile("s_wait_asynccnt 0x0" ::: "memory");
    __syncthreads();
    for (int kcl = 0; kcl < 8; ++kcl) {
      int kc = phase * 8 + kcl;
      // A: this lane's row, interleaved K pattern -> two 32B chunks (full 128B line per lane pair)
      float4 a0 = *(const float4*)(arow + kc * 32 + ka0);
      float4 a1 = *(const float4*)(arow + kc * 32 + ka0 + 4);
      float4 a2 = *(const float4*)(arow + kc * 32 + ka0 + 16);
      float4 a3 = *(const float4*)(arow + kc * 32 + ka0 + 20);
      BfVec av;
      u16x8 u0, u1;
      u0[0]=f2bf(a0.x); u0[1]=f2bf(a0.y); u0[2]=f2bf(a0.z); u0[3]=f2bf(a0.w);
      u0[4]=f2bf(a1.x); u0[5]=f2bf(a1.y); u0[6]=f2bf(a1.z); u0[7]=f2bf(a1.w);
      u1[0]=f2bf(a2.x); u1[1]=f2bf(a2.y); u1[2]=f2bf(a2.z); u1[3]=f2bf(a2.w);
      u1[4]=f2bf(a3.x); u1[5]=f2bf(a3.y); u1[6]=f2bf(a3.z); u1[7]=f2bf(a3.w);
      av.h[0] = u0; av.h[1] = u1;
      #pragma unroll
      for (int t = 0; t < 16; ++t) {
        int n = t * 16 + lm;
        BfVec bv;
        bv.h[0] = *(const u16x8*)(Bsh + n * 256 + kcl * 32 + ka0);
        bv.h[1] = *(const u16x8*)(Bsh + n * 256 + kcl * 32 + ka0 + 16);
        acc[t] = __builtin_amdgcn_wmma_f32_16x16x32_bf16(false, av.v, false, bv.v,
                                                         (short)0, acc[t], false, false);
      }
    }
  }
  // epilogue: + hpart, tanh, scale by v[n], reduce over N
  float p[8];
  #pragma unroll
  for (int j = 0; j < 8; ++j) p[j] = 0.f;
  #pragma unroll
  for (int t = 0; t < 16; ++t) {
    int n = t * 16 + lm;
    float hp = hpart[b * H_ + n];
    float vn = vvec[n];
    #pragma unroll
    for (int j = 0; j < 8; ++j) p[j] += vn * tanhf(acc[t][j] + hp);
  }
  #pragma unroll
  for (int j = 0; j < 8; ++j)
    for (int off = 1; off < 16; off <<= 1) p[j] += __shfl_xor(p[j], off, 32);
  if (lm == 0) {                              // lane0: rows 0..7; lane16: rows 8..15
    #pragma unroll
    for (int j = 0; j < 8; ++j) scores[rowbase + hi * 8 + j] = p[j];
  }
}

// -------- K4: softmax over S per batch ----------
__global__ void softmax_kernel(const float* scores, float* attn) {
  __shared__ float red[256];
  int b = blockIdx.x, tid = threadIdx.x;
  const float* s = scores + b * S_;
  float m = -3.4e38f;
  for (int i = tid; i < S_; i += 256) m = fmaxf(m, s[i]);
  red[tid] = m; __syncthreads();
  for (int off = 128; off > 0; off >>= 1) { if (tid < off) red[tid] = fmaxf(red[tid], red[tid + off]); __syncthreads(); }
  m = red[0]; __syncthreads();
  float sum = 0.f;
  for (int i = tid; i < S_; i += 256) { float e = expf(s[i] - m); attn[b * S_ + i] = e; sum += e; }
  red[tid] = sum; __syncthreads();
  for (int off = 128; off > 0; off >>= 1) { if (tid < off) red[tid] += red[tid + off]; __syncthreads(); }
  float inv = 1.0f / red[0];
  for (int i = tid; i < S_; i += 256) attn[b * S_ + i] *= inv;
}

// -------- K5: context GEMV (enc from L2) + Pgen + cat_project ----------
__global__ void context_kernel(const float* attn, const float* enc, const float* contextW,
                               const float* proj_W, const float* proj_b, const float* h_t,
                               const float* pgen_part, float* catproj, float* Pgen) {
  __shared__ float ash[S_];
  __shared__ float ctx[2 * H_];
  __shared__ float hsh[H_];
  __shared__ float red[256];
  int b = blockIdx.x, tid = threadIdx.x;
  for (int i = tid; i < S_; i += 256) ash[i] = attn[b * S_ + i];
  hsh[tid] = h_t[b * H_ + tid];
  __syncthreads();
  for (int d = tid; d < 2 * H_; d += 256) {
    float s = 0.f;
    const float* e = enc + (size_t)b * S_ * (2 * H_) + d;
    for (int si = 0; si < S_; ++si) s += ash[si] * e[(size_t)si * (2 * H_)];
    ctx[d] = s;
  }
  __syncthreads();
  float pr = ctx[tid] * contextW[tid] + ctx[tid + 256] * contextW[tid + 256];
  red[tid] = pr; __syncthreads();
  for (int off = 128; off > 0; off >>= 1) { if (tid < off) red[tid] += red[tid + off]; __syncthreads(); }
  if (tid == 0) Pgen[b] = pgen_part[b] + red[0];   // reference Pgen is linear (no sigmoid)
  float a = proj_b[tid];
  const float* w = proj_W + (size_t)tid * (3 * H_);
  for (int k = 0; k < 2 * H_; ++k) a += ctx[k] * w[k];
  for (int k = 0; k < H_; ++k)     a += hsh[k] * w[2 * H_ + k];
  catproj[b * H_ + tid] = a;
}

// -------- K6: WMMA vocab GEMM + fused Pgen scale: out[m,n] = Pgen[m]*(cp.outW^T + out_b) ----------
// Block = 256 thr (8 waves); each wave owns one 16-col N-tile, all 8 M-tiles (M=128).
__global__ void __launch_bounds__(256) vocab_kernel(const float* catproj, const float* outW,
                                                    const float* out_b, const float* Pgen,
                                                    float* dout) {
  __shared__ unsigned short Ash[B_ * H_];     // 128x256 bf16 = 64 KB, [m*256 + k]
  __shared__ float Pg[B_];
  int tid = threadIdx.x;
  for (int i = tid; i < B_ * H_; i += 256) Ash[i] = f2bf(catproj[i]);
  if (tid < B_) Pg[tid] = Pgen[tid];
  __syncthreads();
  int wave = tid >> 5, lane = tid & 31;
  int hi = lane >> 4, lm = lane & 15;
  int tile = blockIdx.x * 8 + wave;
  if (tile >= V_ / 16) return;                // V=50000 -> exactly 3125 tiles
  int n = tile * 16 + lm;
  int ka0 = hi ? 8 : 0;
  int kb0 = hi ? 16 : 0;
  v8f vz = {};
  v8f acc[8];
  #pragma unroll
  for (int t = 0; t < 8; ++t) acc[t] = vz;
  const float* wrow = outW + (size_t)n * H_;  // B[k][n] = out_W[n][k]
  for (int kc = 0; kc < 8; ++kc) {
    float4 b0 = *(const float4*)(wrow + kc * 32 + kb0);
    float4 b1 = *(const float4*)(wrow + kc * 32 + kb0 + 4);
    float4 b2 = *(const float4*)(wrow + kc * 32 + kb0 + 8);
    float4 b3 = *(const float4*)(wrow + kc * 32 + kb0 + 12);
    BfVec bv;
    u16x8 u0, u1;
    u0[0]=f2bf(b0.x); u0[1]=f2bf(b0.y); u0[2]=f2bf(b0.z); u0[3]=f2bf(b0.w);
    u0[4]=f2bf(b1.x); u0[5]=f2bf(b1.y); u0[6]=f2bf(b1.z); u0[7]=f2bf(b1.w);
    u1[0]=f2bf(b2.x); u1[1]=f2bf(b2.y); u1[2]=f2bf(b2.z); u1[3]=f2bf(b2.w);
    u1[4]=f2bf(b3.x); u1[5]=f2bf(b3.y); u1[6]=f2bf(b3.z); u1[7]=f2bf(b3.w);
    bv.h[0] = u0; bv.h[1] = u1;
    #pragma unroll
    for (int mt = 0; mt < 8; ++mt) {
      const unsigned short* ar = Ash + (mt * 16 + lm) * H_ + kc * 32;
      BfVec av;
      av.h[0] = *(const u16x8*)(ar + kb0);
      av.h[1] = *(const u16x8*)(ar + kb0 + 8);
      acc[mt] = __builtin_amdgcn_wmma_f32_16x16x32_bf16(false, av.v, false, bv.v,
                                                        (short)0, acc[mt], false, false);
    }
  }
  float ob = out_b[n];
  #pragma unroll
  for (int mt = 0; mt < 8; ++mt) {
    #pragma unroll
    for (int j = 0; j < 8; ++j) {
      int m = mt * 16 + j + 8 * hi;           // C/D layout: VGPR j -> M=j (+8 upper half)
      dout[(size_t)m * VE_ + n] = Pg[m] * (acc[mt][j] + ob);
    }
  }
}

// -------- K7: zero the EXT columns (Pgen*0 base, scatter adds on top) ----------
__global__ void extzero_kernel(float* dout) {
  int b = blockIdx.x, tid = threadIdx.x;
  if (tid < EXT_) dout[(size_t)b * VE_ + V_ + tid] = 0.0f;
}

// -------- K8: scatter-add (1-Pgen)*raw_scores at source_mapping ----------
__global__ void scatter_kernel(const int* srcmap, const float* scores, const float* Pgen,
                               float* dout) {
  int b = blockIdx.x, tid = threadIdx.x;
  float onep = 1.0f - Pgen[b];
  for (int s = tid; s < S_; s += 256) {
    int col = srcmap[b * S_ + s];
    atomicAdd(dout + (size_t)b * VE_ + col, onep * scores[b * S_ + s]);
  }
}

extern "C" void kernel_launch(void* const* d_in, const int* in_sizes, int n_in,
                              void* d_out, int out_size, void* d_ws, size_t ws_size,
                              hipStream_t stream) {
  const int*   word    = (const int*)  d_in[0];
  const float* hidden  = (const float*)d_in[1];
  const float* cell    = (const float*)d_in[2];
  const float* enc     = (const float*)d_in[3];
  const int*   srcmap  = (const int*)  d_in[4];
  // d_in[5]=max_source_len, d_in[6]=max_ext_length (compile-time constants here)
  const float* embed   = (const float*)d_in[7];
  const float* W_ih    = (const float*)d_in[8];
  const float* W_hh    = (const float*)d_in[9];
  const float* b_ih    = (const float*)d_in[10];
  const float* b_hh    = (const float*)d_in[11];
  const float* attn_W  = (const float*)d_in[12];
  const float* attn_b  = (const float*)d_in[13];
  const float* vvec    = (const float*)d_in[14];
  const float* proj_W  = (const float*)d_in[15];
  const float* proj_b  = (const float*)d_in[16];
  const float* out_W   = (const float*)d_in[17];
  const float* out_b   = (const float*)d_in[18];
  const float* contextW= (const float*)d_in[19];
  const float* hiddenW = (const float*)d_in[20];
  const float* wordW   = (const float*)d_in[21];
  const float* bscal   = (const float*)d_in[22];

  float* dout  = (float*)d_out;
  float* h_out = dout + (size_t)B_ * VE_;       // tuple output #2
  float* c_out = h_out + B_ * H_;               // tuple output #3

  float* ws        = (float*)d_ws;
  float* scores_ws = ws;                        // B*S
  float* attnp_ws  = scores_ws + B_ * S_;       // B*S
  float* hpart_ws  = attnp_ws + B_ * S_;        // B*H
  float* catproj_ws= hpart_ws + B_ * H_;        // B*H
  float* pgenp_ws  = catproj_ws + B_ * H_;      // B
  float* pgen_ws   = pgenp_ws + B_;             // B
  unsigned short* webf = (unsigned short*)(pgen_ws + B_);  // 256*512 bf16 (256 KB)

  lstm_kernel   <<<B_, 256, 0, stream>>>(word, hidden, cell, embed, W_ih, W_hh, b_ih, b_hh,
                                         hiddenW, wordW, bscal, pgenp_ws, h_out, c_out);
  hpart_kernel  <<<B_, 256, 0, stream>>>(attn_W, attn_b, h_out, hpart_ws);
  convWe_kernel <<<(NH * KAW) / 256, 256, 0, stream>>>(attn_W, webf);
  scores_kernel <<<(B_ * S_) / 128, 256, 0, stream>>>(enc, webf, hpart_ws, vvec, scores_ws);
  softmax_kernel<<<B_, 256, 0, stream>>>(scores_ws, attnp_ws);
  context_kernel<<<B_, 256, 0, stream>>>(attnp_ws, enc, contextW, proj_W, proj_b, h_out,
                                         pgenp_ws, catproj_ws, pgen_ws);
  vocab_kernel  <<<(V_ / 16 + 7) / 8, 256, 0, stream>>>(catproj_ws, out_W, out_b, pgen_ws, dout);
  extzero_kernel<<<B_, 128, 0, stream>>>(dout);
  scatter_kernel<<<B_, 256, 0, stream>>>(srcmap, scores_ws, pgen_ws, dout);
}
